// MultiHeadAttention_46334107189675
// MI455X (gfx1250) — compile-verified, matrix operand order checked by reading, not run on hardware
//
#include <hip/hip_runtime.h>
#include <hip/hip_bf16.h>

typedef __attribute__((ext_vector_type(16))) _Float16 v16h;
typedef __attribute__((ext_vector_type(8)))  float    v8f;
typedef __attribute__((ext_vector_type(4)))  unsigned int v4u;
typedef __attribute__((ext_vector_type(8)))  unsigned int v8u;
typedef __attribute__((ext_vector_type(4)))  int v4i;

constexpr int Bn = 2, Sn = 2048, En = 1024, Hn = 16, Dn = 64;

#if defined(__AMDGCN__) && __has_builtin(__builtin_amdgcn_global_load_async_to_lds_b128)
#define HAVE_ASYNC_LDS 1
typedef __attribute__((address_space(1))) v4i as1_v4i;
typedef __attribute__((address_space(3))) v4i as3_v4i;
#else
#define HAVE_ASYNC_LDS 0
#endif

// ---------------------------------------------------------------------------
// WMMA helpers (CDNA5 16x16x32 f16 -> f32), layouts per ISA 7.12.2.
// ---------------------------------------------------------------------------
__device__ __forceinline__ v8f wmma16(v16h a, v16h b, v8f c) {
  return __builtin_amdgcn_wmma_f32_16x16x32_f16(
      false, a, false, b, (short)0, c, false, false);
}

template <typename F>
__device__ __forceinline__ v16h load_a16(F&& f) {  // f(m, k) of 16x32 A
  const int lane = threadIdx.x & 31;
  const int m  = lane & 15;
  const int kb = (lane >> 4) << 3;  // 0 or 8
  v16h a;
#pragma unroll
  for (int i = 0; i < 4; ++i) {
    a[2 * i]         = f(m, kb + 2 * i);
    a[2 * i + 1]     = f(m, kb + 2 * i + 1);
    a[8 + 2 * i]     = f(m, 16 + kb + 2 * i);
    a[8 + 2 * i + 1] = f(m, 16 + kb + 2 * i + 1);
  }
  return a;
}

template <typename F>
__device__ __forceinline__ v16h load_b16(F&& f) {  // f(k, n) of 32x16 B
  const int lane = threadIdx.x & 31;
  const int n  = lane & 15;
  const int kb = (lane >> 4) << 3;
  v16h b;
#pragma unroll
  for (int i = 0; i < 4; ++i) {
    b[2 * i]         = f(kb + 2 * i, n);
    b[2 * i + 1]     = f(kb + 2 * i + 1, n);
    b[8 + 2 * i]     = f(16 + kb + 2 * i, n);
    b[8 + 2 * i + 1] = f(16 + kb + 2 * i + 1, n);
  }
  return b;
}

__device__ __forceinline__ void wait_tensorcnt0() {
#if defined(__AMDGCN__) && __has_builtin(__builtin_amdgcn_s_wait_tensorcnt)
  __builtin_amdgcn_s_wait_tensorcnt(0);
#else
  asm volatile("s_wait_tensorcnt 0x0" ::: "memory");
#endif
}

// ---------------------------------------------------------------------------
// TDM: stage one 32-token x 64-d f16 K panel (4 KB) into LDS.
// Hand-built 2D tensor descriptor (D#) per ISA ch.8:
//   group0: count=1 | lds_addr | global_addr[56:0] | type=2
//   group1: data_size=2B, tensor 64 x 2048 (stride 64), tile 64 x 32
// Issued by one wave; completion via TENSORcnt.
// ---------------------------------------------------------------------------
__device__ __forceinline__ void tdm_load_k_panel(const _Float16* gsrc,
                                                 unsigned lds_byte_addr) {
  const unsigned long long ga = (unsigned long long)(size_t)gsrc;
  v4u g0;
  g0[0] = 1u;                                                // count=1
  g0[1] = lds_byte_addr;                                     // lds_addr
  g0[2] = (unsigned)ga;                                      // global_addr lo
  g0[3] = ((unsigned)(ga >> 32) & 0x01FFFFFFu) | (2u << 30); // addr hi | type=2
  v8u g1;
  g1[0] = 1u << 16;                 // data_size = 2 bytes
  g1[1] = (unsigned)Dn << 16;       // tensor_dim0 = 64 (bits 79:48 lo half)
  g1[2] = ((unsigned)Sn & 0xFFFFu) << 16;  // tensor_dim1 = 2048 (lo half)
  g1[3] = (unsigned)Dn << 16;       // tile_dim0 = 64
  g1[4] = 32u;                      // tile_dim1 = 32 (tile_dim2 = 0)
  g1[5] = (unsigned)Dn;             // tensor_dim0_stride = 64
  g1[6] = 0u;
  g1[7] = 0u;
  asm volatile("tensor_load_to_lds %0, %1" :: "s"(g0), "s"(g1) : "memory");
}

// ---------------------------------------------------------------------------
// Kernel 1: fused QKV projection.  Y = X @ W^T + b  (fp32 in, f16 out),
// output layout [B,H,S,D].  128x64 tile / 256-thread workgroup, 8 waves.
// ---------------------------------------------------------------------------
__global__ __launch_bounds__(256) void qkv_proj_kernel(
    const float* __restrict__ Xq, const float* __restrict__ Xk, const float* __restrict__ Xv,
    const float* __restrict__ Wq, const float* __restrict__ Wk, const float* __restrict__ Wv,
    const float* __restrict__ bq, const float* __restrict__ bk, const float* __restrict__ bv,
    _Float16* __restrict__ Qh, _Float16* __restrict__ Kh, _Float16* __restrict__ Vh) {
  const int which = blockIdx.z;
  const float* X    = which == 0 ? Xq : (which == 1 ? Xk : Xv);
  const float* W    = which == 0 ? Wq : (which == 1 ? Wk : Wv);
  const float* bias = which == 0 ? bq : (which == 1 ? bk : bv);
  _Float16* dst     = which == 0 ? Qh : (which == 1 ? Kh : Vh);

  const int f0 = blockIdx.x * 64;
  const int n0 = blockIdx.y * 128;
  const int tid = threadIdx.x, lane = tid & 31, wave = tid >> 5;
  const int wm = (wave & 3) * 32;
  const int wn = (wave >> 2) * 32;

  __shared__ alignas(16) _Float16 As[128][36];
  __shared__ alignas(16) _Float16 Bs[64][36];

  v8f acc[2][2] = {};

  for (int kk = 0; kk < En; kk += 32) {
    {
      const int r = tid >> 1, c = (tid & 1) * 16;
      const float* src = X + (size_t)(n0 + r) * En + kk + c;
      __builtin_prefetch(src + 32, 0, 0);
#pragma unroll
      for (int i = 0; i < 16; ++i) As[r][c + i] = (_Float16)src[i];
    }
    {
      const int r = tid >> 2, c = (tid & 3) * 8;
      const float* src = W + (size_t)(f0 + r) * En + kk + c;
#pragma unroll
      for (int i = 0; i < 8; ++i) Bs[r][c + i] = (_Float16)src[i];
    }
    __syncthreads();

    v16h a0 = load_a16([&](int m, int k) { return As[wm + m][k]; });
    v16h a1 = load_a16([&](int m, int k) { return As[wm + 16 + m][k]; });
#pragma unroll
    for (int j = 0; j < 2; ++j) {
      v16h bm = load_b16([&](int k, int n) { return Bs[wn + j * 16 + n][k]; });
      acc[0][j] = wmma16(a0, bm, acc[0][j]);
      acc[1][j] = wmma16(a1, bm, acc[1][j]);
    }
    __syncthreads();
  }

  const int mh = (lane >> 4) << 3;
  const int n  = lane & 15;
#pragma unroll
  for (int i = 0; i < 2; ++i)
#pragma unroll
    for (int j = 0; j < 2; ++j)
#pragma unroll
      for (int r = 0; r < 8; ++r) {
        const int row = n0 + wm + i * 16 + mh + r;
        const int col = f0 + wn + j * 16 + n;
        const float y = acc[i][j][r] + bias[col];
        const int bi = row / Sn, s = row % Sn;
        const int h = col / Dn, d = col & (Dn - 1);
        dst[(((size_t)bi * Hn + h) * Sn + s) * Dn + d] = (_Float16)y;
      }
}

// ---------------------------------------------------------------------------
// Kernel 2: flash-style softmax statistics (scores never stored).
// Workgroup-cooperative k loop; K panels staged into LDS by the TDM
// (tensor_load_to_lds), double-buffered with s_wait_tensorcnt + barriers.
// ---------------------------------------------------------------------------
__global__ __launch_bounds__(256) void attn_stats_kernel(
    const _Float16* __restrict__ Qh, const _Float16* __restrict__ Kh,
    float2* __restrict__ stats) {
  const int bh = blockIdx.y;
  const int qblk = blockIdx.x * 128;
  const int tid = threadIdx.x, lane = tid & 31, wave = tid >> 5;

  __shared__ alignas(16) _Float16 Qs[128][72];
  __shared__ alignas(16) _Float16 Ks[2][32][64];  // double-buffered 4 KB panels

  {
    const int r = tid >> 1, c = (tid & 1) * 32;
    const _Float16* src = Qh + ((size_t)bh * Sn + qblk + r) * Dn + c;
#pragma unroll
    for (int i = 0; i < 32; ++i) Qs[r][c + i] = src[i];
  }

  const _Float16* Kbase = Kh + (size_t)bh * Sn * Dn;
  const unsigned ldsK0 =
      (unsigned)(unsigned long long)(__attribute__((address_space(3))) void*)&Ks[0][0][0];
  const unsigned ldsK1 =
      (unsigned)(unsigned long long)(__attribute__((address_space(3))) void*)&Ks[1][0][0];

  if (wave == 0) tdm_load_k_panel(Kbase, ldsK0);   // panel 0 in flight
  __syncthreads();                                 // Qs visible

  const int q0 = qblk + wave * 16;
  const int mh = (lane >> 4) << 3;
  const int n  = lane & 15;

  v16h a0 = load_a16([&](int m, int k) { return Qs[wave * 16 + m][k]; });
  v16h a1 = load_a16([&](int m, int k) { return Qs[wave * 16 + m][32 + k]; });

  float M[8], Ssum[8];
#pragma unroll
  for (int r = 0; r < 8; ++r) { M[r] = -1e30f; Ssum[r] = 0.f; }

  const int ktEnd   = (q0 >> 4) + 1;      // this wave's causal extent (16-tiles)
  const int npanels = qblk / 32 + 4;      // workgroup max extent (32-token panels)

  for (int p = 0; p < npanels; ++p) {
    if (wave == 0) wait_tensorcnt0();     // panel p landed in LDS
    __syncthreads();                      // visible to all waves
    if (wave == 0 && p + 1 < npanels)     // prefetch p+1 into the other buffer
      tdm_load_k_panel(Kbase + (size_t)(p + 1) * 32 * Dn, (p & 1) ? ldsK0 : ldsK1);

    const int bufi = p & 1;
#pragma unroll
    for (int sub = 0; sub < 2; ++sub) {
      const int kt = 2 * p + sub;
      if (kt < ktEnd) {
        const int k0 = kt * 16;
        v16h b0 = load_b16([&](int kk, int nn) { return Ks[bufi][sub * 16 + nn][kk]; });
        v16h b1 = load_b16([&](int kk, int nn) { return Ks[bufi][sub * 16 + nn][32 + kk]; });
        v8f c = {};
        c = wmma16(a0, b0, c);
        c = wmma16(a1, b1, c);
#pragma unroll
        for (int r = 0; r < 8; ++r) {
          const int q = q0 + mh + r, k = k0 + n;
          float s = (k <= q) ? c[r] * 0.125f : -1e9f;
          float tmax = s;
#pragma unroll
          for (int off = 8; off >= 1; off >>= 1)
            tmax = fmaxf(tmax, __shfl_xor(tmax, off, 32));
          const float newM = fmaxf(M[r], tmax);
          float pe = __expf(s - newM);
#pragma unroll
          for (int off = 8; off >= 1; off >>= 1)
            pe += __shfl_xor(pe, off, 32);
          Ssum[r] = Ssum[r] * __expf(M[r] - newM) + pe;
          M[r] = newM;
        }
      }
    }
    __syncthreads();                      // everyone done reading buf before reuse
  }

  if (n == 0) {
#pragma unroll
    for (int r = 0; r < 8; ++r)
      stats[(size_t)bh * Sn + q0 + mh + r] = make_float2(M[r], Ssum[r]);
  }
}

// ---------------------------------------------------------------------------
// Kernel 3: recompute scores from LDS-staged K (L2-resident source), normalize
// with stats, write attn exactly once, transpose P via per-wave LDS, WMMA P.V.
// K/V panels staged with GLOBAL_LOAD_ASYNC_TO_LDS_B128 (ASYNCcnt pipelined).
// ---------------------------------------------------------------------------
__global__ __launch_bounds__(256) void softmax_av_kernel(
    const _Float16* __restrict__ Qh, const _Float16* __restrict__ Kh,
    const _Float16* __restrict__ Vh, const float2* __restrict__ stats,
    float* __restrict__ attn, _Float16* __restrict__ Ctx) {
  const int bh = blockIdx.y;
  const int b = bh / Hn, h = bh % Hn;
  const int qblk = blockIdx.x * 128;
  const int tid = threadIdx.x, lane = tid & 31, wave = tid >> 5;

  __shared__ alignas(16) _Float16 Qs[128][72];
  __shared__ alignas(16) _Float16 Ks[2][32][64];
  __shared__ alignas(16) _Float16 Vs[2][32][64];
  __shared__ alignas(16) _Float16 Ps[8][16][40];

  const _Float16* Kbase = Kh + (size_t)bh * Sn * Dn;
  const _Float16* Vbase = Vh + (size_t)bh * Sn * Dn;

  // cooperative async panel stage: 256 threads x 16B = 4 KB per matrix
  auto stage = [&](int p) {
    const _Float16* gK = Kbase + (size_t)p * 32 * Dn + tid * 8;
    const _Float16* gV = Vbase + (size_t)p * 32 * Dn + tid * 8;
    _Float16* lK = &Ks[p & 1][0][0] + tid * 8;
    _Float16* lV = &Vs[p & 1][0][0] + tid * 8;
#if HAVE_ASYNC_LDS
    __builtin_amdgcn_global_load_async_to_lds_b128(
        (as1_v4i*)gK, (as3_v4i*)lK, 0, 0);
    __builtin_amdgcn_global_load_async_to_lds_b128(
        (as1_v4i*)gV, (as3_v4i*)lV, 0, 0);
#else
#pragma unroll
    for (int i = 0; i < 8; ++i) { lK[i] = gK[i]; lV[i] = gV[i]; }
#endif
  };

  {
    const int r = tid >> 1, c = (tid & 1) * 32;
    const _Float16* src = Qh + ((size_t)bh * Sn + qblk + r) * Dn + c;
#pragma unroll
    for (int i = 0; i < 32; ++i) Qs[r][c + i] = src[i];
  }
  stage(0);                               // panel 0 in flight
  __syncthreads();                        // Qs visible

  const int q0 = qblk + wave * 16;
  const int mh = (lane >> 4) << 3;
  const int n  = lane & 15;

  v16h a0 = load_a16([&](int m, int k) { return Qs[wave * 16 + m][k]; });
  v16h a1 = load_a16([&](int m, int k) { return Qs[wave * 16 + m][32 + k]; });

  float M[8], invS[8];
#pragma unroll
  for (int r = 0; r < 8; ++r) {
    const float2 st = stats[(size_t)bh * Sn + q0 + mh + r];
    M[r] = st.x;
    invS[r] = 1.0f / st.y;
  }

  float* arow = attn + ((size_t)bh * Sn + q0) * Sn;
  v8f acc[4] = {};
  const int ktLive  = (q0 >> 4) + 1;
  const int npanels = qblk / 32 + 4;

  for (int p = 0; p < npanels; ++p) {
#if HAVE_ASYNC_LDS
    asm volatile("s_wait_asynccnt 0x0" ::: "memory");  // my share of panel p
#endif
    __syncthreads();                      // panel p fully visible
    if (p + 1 < npanels) stage(p + 1);    // prefetch into other buffer

    const int bufi = p & 1;
    bool any = false;
#pragma unroll
    for (int sub = 0; sub < 2; ++sub) {
      const int kt = 2 * p + sub;
      const int k0 = kt * 16;
      if (kt < ktLive) {
        any = true;
        v16h b0 = load_b16([&](int kk, int nn) { return Ks[bufi][sub * 16 + nn][kk]; });
        v16h b1 = load_b16([&](int kk, int nn) { return Ks[bufi][sub * 16 + nn][32 + kk]; });
        v8f c = {};
        c = wmma16(a0, b0, c);
        c = wmma16(a1, b1, c);
#pragma unroll
        for (int r = 0; r < 8; ++r) {
          const int q = q0 + mh + r, k = k0 + n;
          const float s = (k <= q) ? c[r] * 0.125f : -1e9f;
          const float pe = __expf(s - M[r]) * invS[r];   // masked -> underflow 0
          arow[(size_t)(mh + r) * Sn + k] = pe;
          Ps[wave][mh + r][sub * 16 + n] = (_Float16)pe;
        }
      } else {                             // fully-masked tile: still write zeros
#pragma unroll
        for (int r = 0; r < 8; ++r) {
          arow[(size_t)(mh + r) * Sn + k0 + n] = 0.0f;
          Ps[wave][mh + r][sub * 16 + n] = (_Float16)0.f;
        }
      }
    }
    if (any) {
      asm volatile("s_wait_dscnt 0x0" ::: "memory");     // wave-local transpose
      v16h aP = load_a16([&](int m, int k) { return Ps[wave][m][k]; });
#pragma unroll
      for (int dt = 0; dt < 4; ++dt) {
        v16h bm = load_b16([&](int kk, int nn) { return Vs[bufi][kk][dt * 16 + nn]; });
        acc[dt] = wmma16(aP, bm, acc[dt]);
      }
    }
    __syncthreads();                       // done reading buf before reuse
  }

  // attn tail beyond the workgroup's causal extent is all zeros
  const int ktail = npanels * 32;
#pragma unroll
  for (int r = 0; r < 8; ++r) {
    float* rowp = arow + (size_t)(mh + r) * Sn;
    for (int k = ktail + n * 4; k < Sn; k += 64)
      *(float4*)(rowp + k) = make_float4(0.f, 0.f, 0.f, 0.f);
  }

  // ctx layout [B,S,E] row-major for the output GEMM
#pragma unroll
  for (int dt = 0; dt < 4; ++dt)
#pragma unroll
    for (int r = 0; r < 8; ++r) {
      const size_t row = (size_t)b * Sn + q0 + mh + r;
      Ctx[row * En + h * Dn + dt * 16 + n] = (_Float16)acc[dt][r];
    }
}

// ---------------------------------------------------------------------------
// Kernel 4: output projection  out = ctx(f16) @ Wo^T + bo  (fp32 out)
// ---------------------------------------------------------------------------
__global__ __launch_bounds__(256) void out_proj_kernel(
    const _Float16* __restrict__ Ctx, const float* __restrict__ Wo,
    const float* __restrict__ bo, float* __restrict__ Out) {
  const int f0 = blockIdx.x * 64;
  const int n0 = blockIdx.y * 128;
  const int tid = threadIdx.x, lane = tid & 31, wave = tid >> 5;
  const int wm = (wave & 3) * 32;
  const int wn = (wave >> 2) * 32;

  __shared__ alignas(16) _Float16 As[128][36];
  __shared__ alignas(16) _Float16 Bs[64][36];

  v8f acc[2][2] = {};

  for (int kk = 0; kk < En; kk += 32) {
    {
      const int r = tid >> 1, c = (tid & 1) * 16;
      const _Float16* src = Ctx + (size_t)(n0 + r) * En + kk + c;
#pragma unroll
      for (int i = 0; i < 16; ++i) As[r][c + i] = src[i];
    }
    {
      const int r = tid >> 2, c = (tid & 3) * 8;
      const float* src = Wo + (size_t)(f0 + r) * En + kk + c;
#pragma unroll
      for (int i = 0; i < 8; ++i) Bs[r][c + i] = (_Float16)src[i];
    }
    __syncthreads();

    v16h a0 = load_a16([&](int m, int k) { return As[wm + m][k]; });
    v16h a1 = load_a16([&](int m, int k) { return As[wm + 16 + m][k]; });
#pragma unroll
    for (int j = 0; j < 2; ++j) {
      v16h bm = load_b16([&](int k, int n) { return Bs[wn + j * 16 + n][k]; });
      acc[0][j] = wmma16(a0, bm, acc[0][j]);
      acc[1][j] = wmma16(a1, bm, acc[1][j]);
    }
    __syncthreads();
  }

  const int mh = (lane >> 4) << 3;
  const int n  = lane & 15;
#pragma unroll
  for (int i = 0; i < 2; ++i)
#pragma unroll
    for (int j = 0; j < 2; ++j)
#pragma unroll
      for (int r = 0; r < 8; ++r) {
        const int row = n0 + wm + i * 16 + mh + r;
        const int col = f0 + wn + j * 16 + n;
        Out[(size_t)row * En + col] = acc[i][j][r] + bo[col];
      }
}

// ---------------------------------------------------------------------------
extern "C" void kernel_launch(void* const* d_in, const int* in_sizes, int n_in,
                              void* d_out, int out_size, void* d_ws, size_t ws_size,
                              hipStream_t stream) {
  (void)in_sizes; (void)n_in; (void)out_size; (void)ws_size;
  const float* query = (const float*)d_in[0];
  const float* key_  = (const float*)d_in[1];
  const float* value = (const float*)d_in[2];
  // d_in[3] = mask: causal tril, implemented analytically in-kernel
  const float* Wq = (const float*)d_in[4];
  const float* bq = (const float*)d_in[5];
  const float* Wk = (const float*)d_in[6];
  const float* bk = (const float*)d_in[7];
  const float* Wv = (const float*)d_in[8];
  const float* bv = (const float*)d_in[9];
  const float* Wo = (const float*)d_in[10];
  const float* bo = (const float*)d_in[11];

  float* out  = (float*)d_out;                       // [B,S,E]
  float* attn = out + (size_t)Bn * Sn * En;          // [B,H,S,S]

  _Float16* Qh  = (_Float16*)d_ws;
  _Float16* Kh  = Qh + (size_t)Bn * Sn * En;
  _Float16* Vh  = Kh + (size_t)Bn * Sn * En;
  _Float16* Ctx = Vh + (size_t)Bn * Sn * En;
  float2*   st  = (float2*)(Ctx + (size_t)Bn * Sn * En);

  const dim3 blk(256);
  qkv_proj_kernel<<<dim3(En / 64, (Bn * Sn) / 128, 3), blk, 0, stream>>>(
      query, key_, value, Wq, Wk, Wv, bq, bk, bv, Qh, Kh, Vh);
  attn_stats_kernel<<<dim3(Sn / 128, Bn * Hn), blk, 0, stream>>>(Qh, Kh, st);
  softmax_av_kernel<<<dim3(Sn / 128, Bn * Hn), blk, 0, stream>>>(
      Qh, Kh, Vh, st, attn, Ctx);
  out_proj_kernel<<<dim3(En / 64, (Bn * Sn) / 128), blk, 0, stream>>>(
      Ctx, Wo, bo, out);
}
